// AssociationCortex_5385888989690
// MI455X (gfx1250) — matrix operand order repaired and checked by reading, not compiled, not run on hardware
//
#include <hip/hip_runtime.h>
#include <hip/hip_bf16.h>
#include <math.h>

// ---------------- problem constants ----------------
#define B_SZ  8192
#define D_SZ  1024
#define V_SZ  1024
#define E_SZ  8
#define C_SZ  2048   // D + V
#define ED_SZ 1024
#define OD_SZ 1024
#define TOPK  2
#define FB_STRENGTH 0.5f

typedef __attribute__((ext_vector_type(16))) __bf16 v16bfx;
typedef __attribute__((ext_vector_type(8)))  float  v8f;

// ---------------- helpers ----------------
__device__ __forceinline__ float gelu_exact(float x) {
    return 0.5f * x * (1.0f + erff(x * 0.70710678118654752440f));
}

// Async copy of 32 bytes (two b128 chunks) global -> LDS via the CDNA5
// async-DMA path (tracked by ASYNCcnt). INST_OFFSET applies to both the LDS
// and the global address, so one VGPR pair serves both chunks.
__device__ __forceinline__ void async_copy32(unsigned lds_addr, const __bf16* gptr) {
    unsigned long long ga = (unsigned long long)(size_t)gptr;
    asm volatile("global_load_async_to_lds_b128 %0, %1, off\n\t"
                 "global_load_async_to_lds_b128 %0, %1, off offset:16"
                 :: "v"(lds_addr), "v"(ga) : "memory");
}
__device__ __forceinline__ void wait_async() {
    asm volatile("s_wait_asynccnt 0x0" ::: "memory");
}

// A-fragment (16x32 bf16) from LDS tile with row stride SA (elements).
// Lane l<16: row = base+l, K = 0..15; lane l>=16: row = base+l-16, K = 16..31.
template<int SA>
__device__ __forceinline__ v16bfx load_a_frag(const __bf16* As, int row_base, int lane) {
    int r  = row_base + (lane & 15);
    int kb = (lane < 16) ? 0 : 16;
    const __bf16* p = As + (size_t)r * SA;
    v16bfx a;
#pragma unroll
    for (int j = 0; j < 16; ++j) a[j] = p[kb + j];
    return a;
}

// B-fragment (32x16 bf16) from an N-major (transposed-weight) LDS tile with
// row stride SB. Lane l<16: col = base+l, K = 0..15; lane l>=16: K = 16..31.
// K-contiguous per lane -> pure b128 LDS loads.
template<int SB>
__device__ __forceinline__ v16bfx load_bt_frag(const __bf16* Bs, int col_base, int lane) {
    int n  = col_base + (lane & 15);
    int kb = (lane < 16) ? 0 : 16;
    const __bf16* p = Bs + (size_t)n * SB + kb;
    v16bfx b;
#pragma unroll
    for (int j = 0; j < 16; ++j) b[j] = p[j];
    return b;
}

// ---------------- elementwise / transform kernels ----------------
__global__ __launch_bounds__(256) void cvt_f32_bf16_kernel(const float* __restrict__ src,
                                                           __bf16* __restrict__ dst, size_t n) {
    size_t i = (size_t)blockIdx.x * 256 + threadIdx.x;
    if (i < n) dst[i] = (__bf16)src[i];
}

// src: batch of (R, Cc) f32 ; dst: batch of (Cc, R) bf16 (transposed + converted)
__global__ __launch_bounds__(256) void cvt_transpose_kernel(const float* __restrict__ src,
                                                            __bf16* __restrict__ dst,
                                                            int R, int Cc) {
    __shared__ __bf16 T[32][33];
    const size_t bofs = (size_t)blockIdx.z * R * Cc;
    const int r0 = blockIdx.y * 32, c0 = blockIdx.x * 32;
    const int tc = threadIdx.x & 31, tr = threadIdx.x >> 5;
#pragma unroll
    for (int i = 0; i < 4; ++i) {
        int r = tr + i * 8;
        T[r][tc] = (__bf16)src[bofs + (size_t)(r0 + r) * Cc + c0 + tc];
    }
    __syncthreads();
#pragma unroll
    for (int i = 0; i < 4; ++i) {
        int c = tr + i * 8;
        dst[bofs + (size_t)(c0 + c) * R + r0 + tc] = T[tc][c];
    }
}

__global__ __launch_bounds__(256) void concat_cvt_kernel(const float* __restrict__ dorsal,
                                                         const float* __restrict__ ventral,
                                                         __bf16* __restrict__ xb) {
    size_t i = (size_t)blockIdx.x * 256 + threadIdx.x;
    size_t n = (size_t)B_SZ * C_SZ;
    if (i < n) {
        size_t r = i / C_SZ, c = i % C_SZ;
        float v = (c < D_SZ) ? dorsal[r * D_SZ + c] : ventral[r * V_SZ + (c - D_SZ)];
        xb[i] = (__bf16)v;
    }
}

__global__ __launch_bounds__(256) void zero_init_kernel(float* __restrict__ bound,
                                                        int* __restrict__ cnt, size_t n) {
    size_t i = (size_t)blockIdx.x * 256 + threadIdx.x;
    if (i < n) bound[i] = 0.0f;
    if (i < E_SZ) cnt[i] = 0;
}

// ---------------- gate: logits -> top2 softmax -> bucket lists ----------------
__global__ __launch_bounds__(256) void gate_kernel(const float* __restrict__ dorsal,
                                                   const float* __restrict__ ventral,
                                                   const float* __restrict__ gate_W,
                                                   float* __restrict__ gw_out,
                                                   int* __restrict__ rows_e,
                                                   float* __restrict__ wts_e,
                                                   int* __restrict__ cnt)
{
    const int r = blockIdx.x;
    const int tid = threadIdx.x;
    float acc[E_SZ] = {};
    for (int c = tid; c < C_SZ; c += 256) {
        float xv = (c < D_SZ) ? dorsal[(size_t)r * D_SZ + c]
                              : ventral[(size_t)r * V_SZ + (c - D_SZ)];
#pragma unroll
        for (int e = 0; e < E_SZ; ++e) acc[e] += xv * gate_W[(size_t)c * E_SZ + e];
    }
    __shared__ float s[E_SZ * 256];
#pragma unroll
    for (int e = 0; e < E_SZ; ++e) s[e * 256 + tid] = acc[e];
    __syncthreads();
    for (int off = 128; off > 0; off >>= 1) {
        if (tid < off) {
#pragma unroll
            for (int e = 0; e < E_SZ; ++e) s[e * 256 + tid] += s[e * 256 + tid + off];
        }
        __syncthreads();
    }
    if (tid == 0) {
        float v[E_SZ];
#pragma unroll
        for (int e = 0; e < E_SZ; ++e) v[e] = s[e * 256];
        int i0 = 0;
#pragma unroll
        for (int e = 1; e < E_SZ; ++e) if (v[e] > v[i0]) i0 = e;
        int i1 = (i0 == 0) ? 1 : 0;
#pragma unroll
        for (int e = 0; e < E_SZ; ++e) if (e != i0 && v[e] > v[i1]) i1 = e;
        float e1  = expf(v[i1] - v[i0]);
        float inv = 1.0f / (1.0f + e1);
        float w0 = inv, w1 = e1 * inv;
        float* g = gw_out + (size_t)r * E_SZ;
#pragma unroll
        for (int e = 0; e < E_SZ; ++e) g[e] = 0.0f;
        g[i0] = w0; g[i1] = w1;
        int p0 = atomicAdd(&cnt[i0], 1);
        rows_e[i0 * B_SZ + p0] = r; wts_e[i0 * B_SZ + p0] = w0;
        int p1 = atomicAdd(&cnt[i1], 1);
        rows_e[i1 * B_SZ + p1] = r; wts_e[i1 * B_SZ + p1] = w1;
    }
}

__global__ void prefix_kernel(const int* __restrict__ cnt, int* __restrict__ offs) {
    if (threadIdx.x == 0 && blockIdx.x == 0) {
        int a = 0;
#pragma unroll
        for (int e = 0; e < E_SZ; ++e) { offs[e] = a; a += cnt[e]; }
    }
}

// ======================= WMMA GEMM cores =======================
// Block = 256 threads (8 waves); block tile 128x128; wave tile 64x32; K-tile 32.
// LDS double-buffered; both A and B(transposed weights) tiles are [128][40] bf16
// (row stride 80 B = 5*16 B: 16B-aligned chunks, bank-spread).

#define LDSTRIDE 40

// ---------------- expert layer 1: h = gelu(x_gathered @ W1[e] + b1[e]) ----------------
__global__ __launch_bounds__(256) void expert_l1_kernel(const __bf16* __restrict__ xb,   // (B,C)
                                                        const __bf16* __restrict__ W1t,  // (E,ED,C)
                                                        const float*  __restrict__ b1,   // (E,ED)
                                                        const int*    __restrict__ rows_e,
                                                        const int*    __restrict__ cnt,
                                                        const int*    __restrict__ offs,
                                                        __bf16* __restrict__ h)          // (B*TOPK, ED)
{
    const int e   = blockIdx.z;
    const int n_e = cnt[e];
    const int m0  = blockIdx.y * 128;
    if (m0 >= n_e) return;
    const int n0  = blockIdx.x * 128;

    __shared__ __bf16 As[2][128][LDSTRIDE];
    __shared__ __bf16 Bs[2][128][LDSTRIDE];

    const int tid = threadIdx.x, lane = tid & 31, wave = tid >> 5;
    const int mo = (wave >> 2) * 64, no = (wave & 3) * 32;
    const int srow = tid >> 1, scb = (tid & 1) * 16;   // staging: row 0..127, 16-elem chunk

    // clamped gather row (rows beyond n_e only feed never-stored outputs)
    const int apos  = m0 + srow;
    const int aposc = (apos < n_e) ? apos : (n_e - 1);
    const int agr   = rows_e[e * B_SZ + aposc];
    const __bf16* arowp = xb + (size_t)agr * C_SZ + scb;
    const __bf16* browp = W1t + (size_t)e * ED_SZ * C_SZ + (size_t)(n0 + srow) * C_SZ + scb;

    v8f acc[4][2] = {};
    const int NK = C_SZ / 32;

    // prologue: stage tile 0 into buffer 0
    async_copy32((unsigned)(size_t)&As[0][srow][scb], arowp);
    async_copy32((unsigned)(size_t)&Bs[0][srow][scb], browp);

    for (int kt = 0; kt < NK; ++kt) {
        const int buf = kt & 1;
        wait_async();
        __syncthreads();
        if (kt + 1 < NK) {
            async_copy32((unsigned)(size_t)&As[buf ^ 1][srow][scb], arowp + (kt + 1) * 32);
            async_copy32((unsigned)(size_t)&Bs[buf ^ 1][srow][scb], browp + (kt + 1) * 32);
        }
        v16bfx bf0 = load_bt_frag<LDSTRIDE>(&Bs[buf][0][0], no, lane);
        v16bfx bf1 = load_bt_frag<LDSTRIDE>(&Bs[buf][0][0], no + 16, lane);
#pragma unroll
        for (int im = 0; im < 4; ++im) {
            v16bfx af = load_a_frag<LDSTRIDE>(&As[buf][0][0], mo + im * 16, lane);
            acc[im][0] = __builtin_amdgcn_wmma_f32_16x16x32_bf16(false, af, false, bf0, (short)0, acc[im][0], false, false);
            acc[im][1] = __builtin_amdgcn_wmma_f32_16x16x32_bf16(false, af, false, bf1, (short)0, acc[im][1], false, false);
        }
    }

    const int hi8 = (lane >= 16) ? 8 : 0;
    const int hb  = offs[e];
#pragma unroll
    for (int im = 0; im < 4; ++im)
#pragma unroll
        for (int in = 0; in < 2; ++in) {
            int col = n0 + no + in * 16 + (lane & 15);
            int rl  = m0 + mo + im * 16 + hi8;
            float bias = b1[(size_t)e * ED_SZ + col];
#pragma unroll
            for (int i = 0; i < 8; ++i) {
                int row_local = rl + i;
                if (row_local < n_e)
                    h[(size_t)(hb + row_local) * ED_SZ + col] =
                        (__bf16)gelu_exact(acc[im][in][i] + bias);
            }
        }
}

// ---------------- expert layer 2: bound += w * (h @ W2[e] + b2[e]) ----------------
__global__ __launch_bounds__(256) void expert_l2_kernel(const __bf16* __restrict__ h,    // (B*TOPK, ED)
                                                        const __bf16* __restrict__ W2t,  // (E,ED,ED) transposed
                                                        const float*  __restrict__ b2,   // (E,ED)
                                                        const int*    __restrict__ rows_e,
                                                        const float*  __restrict__ wts_e,
                                                        const int*    __restrict__ cnt,
                                                        const int*    __restrict__ offs,
                                                        float* __restrict__ bound)       // (B,ED)
{
    const int e   = blockIdx.z;
    const int n_e = cnt[e];
    const int m0  = blockIdx.y * 128;
    if (m0 >= n_e) return;
    const int n0  = blockIdx.x * 128;

    __shared__ __bf16 As[2][128][LDSTRIDE];
    __shared__ __bf16 Bs[2][128][LDSTRIDE];

    const int tid = threadIdx.x, lane = tid & 31, wave = tid >> 5;
    const int mo = (wave >> 2) * 64, no = (wave & 3) * 32;
    const int srow = tid >> 1, scb = (tid & 1) * 16;

    const int hb    = offs[e];
    const int apos  = m0 + srow;
    const int aposc = (apos < n_e) ? apos : (n_e - 1);
    const __bf16* arowp = h + (size_t)(hb + aposc) * ED_SZ + scb;
    const __bf16* browp = W2t + (size_t)e * ED_SZ * ED_SZ + (size_t)(n0 + srow) * ED_SZ + scb;

    v8f acc[4][2] = {};
    const int NK = ED_SZ / 32;

    async_copy32((unsigned)(size_t)&As[0][srow][scb], arowp);
    async_copy32((unsigned)(size_t)&Bs[0][srow][scb], browp);

    for (int kt = 0; kt < NK; ++kt) {
        const int buf = kt & 1;
        wait_async();
        __syncthreads();
        if (kt + 1 < NK) {
            async_copy32((unsigned)(size_t)&As[buf ^ 1][srow][scb], arowp + (kt + 1) * 32);
            async_copy32((unsigned)(size_t)&Bs[buf ^ 1][srow][scb], browp + (kt + 1) * 32);
        }
        v16bfx bf0 = load_bt_frag<LDSTRIDE>(&Bs[buf][0][0], no, lane);
        v16bfx bf1 = load_bt_frag<LDSTRIDE>(&Bs[buf][0][0], no + 16, lane);
#pragma unroll
        for (int im = 0; im < 4; ++im) {
            v16bfx af = load_a_frag<LDSTRIDE>(&As[buf][0][0], mo + im * 16, lane);
            acc[im][0] = __builtin_amdgcn_wmma_f32_16x16x32_bf16(false, af, false, bf0, (short)0, acc[im][0], false, false);
            acc[im][1] = __builtin_amdgcn_wmma_f32_16x16x32_bf16(false, af, false, bf1, (short)0, acc[im][1], false, false);
        }
    }

    const int hi8 = (lane >= 16) ? 8 : 0;
#pragma unroll
    for (int im = 0; im < 4; ++im)
#pragma unroll
        for (int in = 0; in < 2; ++in) {
            int col = n0 + no + in * 16 + (lane & 15);
            int rl  = m0 + mo + im * 16 + hi8;
            float bias = b2[(size_t)e * ED_SZ + col];
#pragma unroll
            for (int i = 0; i < 8; ++i) {
                int row_local = rl + i;
                if (row_local < n_e) {
                    float w  = wts_e[e * B_SZ + row_local];
                    int   gr = rows_e[e * B_SZ + row_local];
                    unsafeAtomicAdd(&bound[(size_t)gr * ED_SZ + col],
                                    w * (acc[im][in][i] + bias));
                }
            }
        }
}

// ---------------- generic GEMM: out = scale*(A@Bw) + bias (Bw pre-transposed N,K) ----------------
__global__ __launch_bounds__(256) void gemm_bf16_kernel(const __bf16* __restrict__ A,   // (M,Kd)
                                                        const __bf16* __restrict__ Bt,  // (N,Kd)
                                                        const float*  __restrict__ bias,// (N) or null
                                                        float scale,
                                                        float* __restrict__ out,        // (M,N)
                                                        int N, int Kd)
{
    const int n0 = blockIdx.x * 128;
    const int m0 = blockIdx.y * 128;

    __shared__ __bf16 As[2][128][LDSTRIDE];
    __shared__ __bf16 Bs[2][128][LDSTRIDE];

    const int tid = threadIdx.x, lane = tid & 31, wave = tid >> 5;
    const int mo = (wave >> 2) * 64, no = (wave & 3) * 32;
    const int srow = tid >> 1, scb = (tid & 1) * 16;

    const __bf16* arowp = A  + (size_t)(m0 + srow) * Kd + scb;
    const __bf16* browp = Bt + (size_t)(n0 + srow) * Kd + scb;

    v8f acc[4][2] = {};
    const int NK = Kd / 32;

    async_copy32((unsigned)(size_t)&As[0][srow][scb], arowp);
    async_copy32((unsigned)(size_t)&Bs[0][srow][scb], browp);

    for (int kt = 0; kt < NK; ++kt) {
        const int buf = kt & 1;
        wait_async();
        __syncthreads();
        if (kt + 1 < NK) {
            async_copy32((unsigned)(size_t)&As[buf ^ 1][srow][scb], arowp + (kt + 1) * 32);
            async_copy32((unsigned)(size_t)&Bs[buf ^ 1][srow][scb], browp + (kt + 1) * 32);
        }
        v16bfx bf0 = load_bt_frag<LDSTRIDE>(&Bs[buf][0][0], no, lane);
        v16bfx bf1 = load_bt_frag<LDSTRIDE>(&Bs[buf][0][0], no + 16, lane);
#pragma unroll
        for (int im = 0; im < 4; ++im) {
            v16bfx af = load_a_frag<LDSTRIDE>(&As[buf][0][0], mo + im * 16, lane);
            acc[im][0] = __builtin_amdgcn_wmma_f32_16x16x32_bf16(false, af, false, bf0, (short)0, acc[im][0], false, false);
            acc[im][1] = __builtin_amdgcn_wmma_f32_16x16x32_bf16(false, af, false, bf1, (short)0, acc[im][1], false, false);
        }
    }

    const int hi8 = (lane >= 16) ? 8 : 0;
#pragma unroll
    for (int im = 0; im < 4; ++im)
#pragma unroll
        for (int in = 0; in < 2; ++in) {
            int col = n0 + no + in * 16 + (lane & 15);
            int rb  = m0 + mo + im * 16 + hi8;
            float bv = bias ? bias[col] : 0.0f;
#pragma unroll
            for (int i = 0; i < 8; ++i)
                out[(size_t)(rb + i) * N + col] = scale * acc[im][in][i] + bv;
        }
}

// ---------------- host launcher ----------------
extern "C" void kernel_launch(void* const* d_in, const int* in_sizes, int n_in,
                              void* d_out, int out_size, void* d_ws, size_t ws_size,
                              hipStream_t stream) {
    const float* dorsal  = (const float*)d_in[0];
    const float* ventral = (const float*)d_in[1];
    const float* gate_W  = (const float*)d_in[2];
    const float* W1      = (const float*)d_in[3];
    const float* b1      = (const float*)d_in[4];
    const float* W2      = (const float*)d_in[5];
    const float* b2      = (const float*)d_in[6];
    const float* Wo      = (const float*)d_in[7];
    const float* bo      = (const float*)d_in[8];
    const float* Wfd     = (const float*)d_in[9];
    const float* Wfv     = (const float*)d_in[10];

    float* out_assoc = (float*)d_out;                                   // (B,OD)
    float* out_fbd   = out_assoc + (size_t)B_SZ * OD_SZ;                // (B,D)
    float* out_fbv   = out_fbd   + (size_t)B_SZ * D_SZ;                 // (B,V)
    float* out_gw    = out_fbv   + (size_t)B_SZ * V_SZ;                 // (B,E)

    // workspace carve-up (256B aligned)
    char* ws = (char*)d_ws;
    size_t off = 0;
    auto alloc = [&](size_t bytes) -> char* {
        char* p = ws + off;
        off += (bytes + 255) & ~(size_t)255;
        return p;
    };
    __bf16* xb     = (__bf16*)alloc((size_t)B_SZ * C_SZ * 2);
    __bf16* W1t    = (__bf16*)alloc((size_t)E_SZ * C_SZ * ED_SZ * 2);   // (E,ED,C)
    __bf16* W2t    = (__bf16*)alloc((size_t)E_SZ * ED_SZ * ED_SZ * 2);  // (E,ED,ED) T
    __bf16* Wot    = (__bf16*)alloc((size_t)ED_SZ * OD_SZ * 2);         // (OD,ED)
    __bf16* Wfdt   = (__bf16*)alloc((size_t)OD_SZ * D_SZ * 2);          // (D,OD)
    __bf16* Wfvt   = (__bf16*)alloc((size_t)OD_SZ * V_SZ * 2);          // (V,OD)
    __bf16* hbuf   = (__bf16*)alloc((size_t)B_SZ * TOPK * ED_SZ * 2);
    float*  bound  = (float*) alloc((size_t)B_SZ * ED_SZ * 4);
    __bf16* boundb = (__bf16*)alloc((size_t)B_SZ * ED_SZ * 2);
    __bf16* assocb = (__bf16*)alloc((size_t)B_SZ * OD_SZ * 2);
    int*    rows_e = (int*)   alloc((size_t)E_SZ * B_SZ * 4);
    float*  wts_e  = (float*) alloc((size_t)E_SZ * B_SZ * 4);
    int*    cnt    = (int*)   alloc(E_SZ * 4);
    int*    offs   = (int*)   alloc(E_SZ * 4);
    (void)ws_size; (void)in_sizes; (void)n_in; (void)out_size;

    auto blocks = [](size_t n) { return (unsigned)((n + 255) / 256); };

    // 0) zero accumulators + counters
    zero_init_kernel<<<blocks((size_t)B_SZ * ED_SZ), 256, 0, stream>>>(bound, cnt, (size_t)B_SZ * ED_SZ);

    // 1) weight conversion + transpose to (N,K) bf16; activations to bf16
    cvt_transpose_kernel<<<dim3(ED_SZ / 32, C_SZ / 32, E_SZ), 256, 0, stream>>>(W1, W1t, C_SZ, ED_SZ);
    cvt_transpose_kernel<<<dim3(ED_SZ / 32, ED_SZ / 32, E_SZ), 256, 0, stream>>>(W2, W2t, ED_SZ, ED_SZ);
    cvt_transpose_kernel<<<dim3(OD_SZ / 32, ED_SZ / 32, 1), 256, 0, stream>>>(Wo, Wot, ED_SZ, OD_SZ);
    cvt_transpose_kernel<<<dim3(D_SZ / 32, OD_SZ / 32, 1), 256, 0, stream>>>(Wfd, Wfdt, OD_SZ, D_SZ);
    cvt_transpose_kernel<<<dim3(V_SZ / 32, OD_SZ / 32, 1), 256, 0, stream>>>(Wfv, Wfvt, OD_SZ, V_SZ);
    concat_cvt_kernel<<<blocks((size_t)B_SZ * C_SZ), 256, 0, stream>>>(dorsal, ventral, xb);

    // 2) gate: logits, top-2 softmax (writes gate_weights output), expert buckets
    gate_kernel<<<B_SZ, 256, 0, stream>>>(dorsal, ventral, gate_W, out_gw, rows_e, wts_e, cnt);
    prefix_kernel<<<1, 32, 0, stream>>>(cnt, offs);

    // 3) sparse expert FFN (top-2 only)
    dim3 egrid(ED_SZ / 128, B_SZ / 128, E_SZ);
    expert_l1_kernel<<<egrid, 256, 0, stream>>>(xb, W1t, b1, rows_e, cnt, offs, hbuf);
    expert_l2_kernel<<<egrid, 256, 0, stream>>>(hbuf, W2t, b2, rows_e, wts_e, cnt, offs, bound);

    // 4) assoc = bound @ Wo + bo
    cvt_f32_bf16_kernel<<<blocks((size_t)B_SZ * ED_SZ), 256, 0, stream>>>(bound, boundb, (size_t)B_SZ * ED_SZ);
    gemm_bf16_kernel<<<dim3(OD_SZ / 128, B_SZ / 128), 256, 0, stream>>>(boundb, Wot, bo, 1.0f, out_assoc, OD_SZ, ED_SZ);

    // 5) feedback projections
    cvt_f32_bf16_kernel<<<blocks((size_t)B_SZ * OD_SZ), 256, 0, stream>>>(out_assoc, assocb, (size_t)B_SZ * OD_SZ);
    gemm_bf16_kernel<<<dim3(D_SZ / 128, B_SZ / 128), 256, 0, stream>>>(assocb, Wfdt, nullptr, FB_STRENGTH, out_fbd, D_SZ, OD_SZ);
    gemm_bf16_kernel<<<dim3(V_SZ / 128, B_SZ / 128), 256, 0, stream>>>(assocb, Wfvt, nullptr, FB_STRENGTH, out_fbv, V_SZ, OD_SZ);
}
